// VanillaRNN_50130858279289
// MI455X (gfx1250) — matrix operand order repaired and checked by reading, not compiled, not run on hardware
//
#include <hip/hip_runtime.h>
#include <hip/hip_bf16.h>

// ---------------------------------------------------------------------------
// 2-layer vanilla RNN on MI455X (gfx1250, wave32, WMMA).
//   8 persistent workgroups (one per 16-row batch tile), each loops SEQ steps.
//   f16 WMMA (v_wmma_f32_16x16x32_f16), f32 accumulation.
//   Weights pre-packed into per-lane WMMA B-fragment order in d_ws (L2-resident).
//   h0/h1 live in LDS as the WMMA A-matrices [x|h0] and [h0|h1].
// ---------------------------------------------------------------------------

typedef __attribute__((ext_vector_type(16))) _Float16 v16h;
typedef __attribute__((ext_vector_type(8)))  _Float16 v8h;
typedef __attribute__((ext_vector_type(8)))  float    v8f;

#define VOCAB  256
#define EMBED  256
#define HIDDEN 512
#define BATCH  128
#define SEQ    1024

// packed-weight region sizes (in halfs)
#define B0_HALFS (32 * 24 * 512)   // layer0: N=512 (32 ntiles), K=768 (24 ksteps)
#define B1_HALFS (32 * 32 * 512)   // layer1: N=512, K=1024
#define B2_HALFS (16 * 16 * 512)   // out:    N=256, K=512
#define B1_OFF   (B0_HALFS)
#define B2_OFF   (B0_HALFS + B1_HALFS)
#define WP_TOTAL (B0_HALFS + B1_HALFS + B2_HALFS)   // 1,048,576 halfs = 2 MB
#define BIAS_N   (512 + 512 + 256)                  // 1280 f32 appended after Wp

// LDS A-matrix row strides (halfs), padded +8 halfs (16B) to spread banks
#define S0 (768 + 8)
#define S1 (1024 + 8)

__device__ __forceinline__ v8f wmma_f16(v16h a, v16h b, v8f c) {
  // (neg_a, A, neg_b, B, c_mod, C, reuse_a, reuse_b)
  return __builtin_amdgcn_wmma_f32_16x16x32_f16(false, a, false, b, (short)0, c,
                                                false, false);
}

// A-fragment (16x32 f16) from a row-major LDS row.
// Layout (matches packed B): lane<16 holds K = kbase+{0..15}; lane>=16 holds
// K = kbase+{16..31}.  hi8 = (lane>>4)*8.
__device__ __forceinline__ v16h load_a_lds(const _Float16* row, int kbase, int hi8) {
  v8h lo = *(const v8h*)(row + kbase + 2 * hi8);        // 16B ds_load_b128
  v8h hh = *(const v8h*)(row + kbase + 2 * hi8 + 8);    // 16B ds_load_b128
  return __builtin_shufflevector(lo, hh, 0, 1, 2, 3, 4, 5, 6, 7,
                                 8, 9, 10, 11, 12, 13, 14, 15);
}

// Scatter an f16 D-tile (this lane's 8 rows) into a row-major LDS A buffer.
// C/D layout: lane -> N = lane&15, rows M = m0..m0+7 with m0 = (lane>>4)*8.
__device__ __forceinline__ void store_h_lds(_Float16* colbase, int stride,
                                            const _Float16 h[8], int m0, int n) {
#pragma unroll
  for (int j = 0; j < 8; ++j) colbase[(m0 + j) * stride + n] = h[j];
}

// Branchless, overflow-safe tanh using the v_exp_f32 trans op.
__device__ __forceinline__ float fast_tanh(float x) {
  float ax = __builtin_fabsf(x);
  float e  = __expf(-2.0f * ax);          // e in (0,1]
  float r  = (1.0f - e) / (1.0f + e);
  return __builtin_copysignf(r, x);
}

// ---------------------------------------------------------------------------
// Pack kernel: f32 weights -> f16 WMMA B fragments in lane order, plus fused
// biases.  B element (nt, kk, lane, i) holds W[n][k] with n = nt*16 + (lane&15),
// k = kk*32 + (lane>>4)*16 + i  (this code's contiguous k order: lanes 0-15 K=0..15,
// lanes 16-31 K=16..31, column = lane&15).
// ---------------------------------------------------------------------------
__global__ __launch_bounds__(256) void rnn_pack_kernel(
    const float* __restrict__ Wih0, const float* __restrict__ Whh0,
    const float* __restrict__ Wih1, const float* __restrict__ Whh1,
    const float* __restrict__ Wout,
    const float* __restrict__ bih0, const float* __restrict__ bhh0,
    const float* __restrict__ bih1, const float* __restrict__ bhh1,
    const float* __restrict__ bout,
    _Float16* __restrict__ Wp, float* __restrict__ bias) {
  int idx = blockIdx.x * 256 + threadIdx.x;
  if (idx < B0_HALFS) {
    int rel = idx;
    int i = rel & 15, l = (rel >> 4) & 31, q = rel >> 9;
    int kk = q % 24, nt = q / 24;
    int n = nt * 16 + (l & 15);
    int k = kk * 32 + ((l >> 4) << 4) + i;
    float v = (k < EMBED) ? Wih0[n * EMBED + k] : Whh0[n * HIDDEN + (k - EMBED)];
    Wp[idx] = (_Float16)v;
  } else if (idx < B2_OFF) {
    int rel = idx - B1_OFF;
    int i = rel & 15, l = (rel >> 4) & 31, q = rel >> 9;
    int kk = q % 32, nt = q / 32;
    int n = nt * 16 + (l & 15);
    int k = kk * 32 + ((l >> 4) << 4) + i;
    float v = (k < HIDDEN) ? Wih1[n * HIDDEN + k] : Whh1[n * HIDDEN + (k - HIDDEN)];
    Wp[idx] = (_Float16)v;
  } else if (idx < WP_TOTAL) {
    int rel = idx - B2_OFF;
    int i = rel & 15, l = (rel >> 4) & 31, q = rel >> 9;
    int kk = q % 16, nt = q / 16;
    int n = nt * 16 + (l & 15);
    int k = kk * 32 + ((l >> 4) << 4) + i;
    Wp[idx] = (_Float16)Wout[n * HIDDEN + k];
  } else if (idx < WP_TOTAL + BIAS_N) {
    int b = idx - WP_TOTAL;
    if (b < 512)       bias[b] = bih0[b] + bhh0[b];
    else if (b < 1024) bias[b] = bih1[b - 512] + bhh1[b - 512];
    else               bias[b] = bout[b - 1024];
  }
}

// ---------------------------------------------------------------------------
// Persistent RNN kernel: one block per 16 batch rows, 16 waves (512 threads).
// Wave w owns hidden N-tiles {2w, 2w+1} for layers 0/1 and vocab N-tile w.
// ---------------------------------------------------------------------------
__global__ __launch_bounds__(512, 1) void rnn_persistent_kernel(
    const int* __restrict__ x, const float* __restrict__ emb,
    const _Float16* __restrict__ Wp,
    const float* __restrict__ bias0, const float* __restrict__ bias1,
    const float* __restrict__ bias2,
    float* __restrict__ out) {
  __shared__ __align__(16) _Float16 A0[16 * S0];  // [x_t (256) | h0 (512)]
  __shared__ __align__(16) _Float16 A1[16 * S1];  // [h0_new (512) | h1 (512)]

  const int tid    = threadIdx.x;
  const int lane   = tid & 31;
  const int w      = tid >> 5;        // wave 0..15
  const int lrow   = lane & 15;       // A row / C column within tile
  const int hi8    = (lane >> 4) << 3;
  const int m0     = (lane >> 4) << 3;
  const int wgbase = blockIdx.x * 16; // first batch row of this block

  // h0 = h1 = 0 at t=0
  for (int i = tid; i < 16 * S0; i += 512) A0[i] = (_Float16)0.0f;
  for (int i = tid; i < 16 * S1; i += 512) A1[i] = (_Float16)0.0f;

  const int nt0 = 2 * w, nt1 = 2 * w + 1;
  const float bL0a = bias0[nt0 * 16 + lrow];
  const float bL0b = bias0[nt1 * 16 + lrow];
  const float bL1a = bias1[nt0 * 16 + lrow];
  const float bL1b = bias1[nt1 * 16 + lrow];
  const float bO   = bias2[w * 16 + lrow];

  const _Float16* B0a = Wp + (size_t)(nt0 * 24) * 512 + lane * 16;
  const _Float16* B0b = Wp + (size_t)(nt1 * 24) * 512 + lane * 16;
  const _Float16* B1a = Wp + B1_OFF + (size_t)(nt0 * 32) * 512 + lane * 16;
  const _Float16* B1b = Wp + B1_OFF + (size_t)(nt1 * 32) * 512 + lane * 16;
  const _Float16* B2p = Wp + B2_OFF + (size_t)(w * 16) * 512 + lane * 16;

  const _Float16* arow0 = A0 + lrow * S0;
  const _Float16* arow1 = A1 + lrow * S1;
  const int xrow = (wgbase + w) * SEQ;

  __syncthreads();

  for (int t = 0; t < SEQ; ++t) {
    // ---- embedding gather: wave w fills row w, lane covers 8 columns ----
    {
      int idx = x[xrow + t];
      const float4* s = (const float4*)(emb + (size_t)idx * EMBED + lane * 8);
      float4 v0 = s[0], v1 = s[1];
      _Float16* d = A0 + w * S0 + lane * 8;
      d[0] = (_Float16)v0.x; d[1] = (_Float16)v0.y;
      d[2] = (_Float16)v0.z; d[3] = (_Float16)v0.w;
      d[4] = (_Float16)v1.x; d[5] = (_Float16)v1.y;
      d[6] = (_Float16)v1.z; d[7] = (_Float16)v1.w;
    }
    __syncthreads();  // B1: x_t ready, prev-iter A0 h-slot writes visible

    // ---- layer 0: [x|h0] (16x768) @ W0^T, 24 k-steps, 2 N-tiles/wave ----
    v8f c0, c1;
#pragma unroll
    for (int j = 0; j < 8; ++j) { c0[j] = bL0a; c1[j] = bL0b; }
    {
      const _Float16* p0 = B0a;
      const _Float16* p1 = B0b;
#pragma unroll 4
      for (int kk = 0; kk < 24; ++kk) {
        v16h a  = load_a_lds(arow0, kk * 32, hi8);
        v16h b0 = *(const v16h*)p0;   // 2x global_load_b128, L2-resident
        v16h b1 = *(const v16h*)p1;
        c0 = wmma_f16(a, b0, c0);
        c1 = wmma_f16(a, b1, c1);
        p0 += 512; p1 += 512;
      }
    }
    _Float16 h0a[8], h0b[8];
#pragma unroll
    for (int j = 0; j < 8; ++j) {
      h0a[j] = (_Float16)fast_tanh(c0[j]);
      h0b[j] = (_Float16)fast_tanh(c1[j]);
    }
    // h0_new -> A1[:,0:512] (nobody reads A1 h0-slot until after B2)
    store_h_lds(A1 + nt0 * 16, S1, h0a, m0, lrow);
    store_h_lds(A1 + nt1 * 16, S1, h0b, m0, lrow);
    __syncthreads();  // B2: h0_new visible; all layer-0 reads of A0 done
    // h0_new -> A0 recurrent slot for the NEXT timestep (layer1 ignores A0)
    store_h_lds(A0 + EMBED + nt0 * 16, S0, h0a, m0, lrow);
    store_h_lds(A0 + EMBED + nt1 * 16, S0, h0b, m0, lrow);

    // ---- layer 1: [h0_new|h1] (16x1024) @ W1^T, 32 k-steps ----
#pragma unroll
    for (int j = 0; j < 8; ++j) { c0[j] = bL1a; c1[j] = bL1b; }
    {
      const _Float16* p0 = B1a;
      const _Float16* p1 = B1b;
#pragma unroll 4
      for (int kk = 0; kk < 32; ++kk) {
        v16h a  = load_a_lds(arow1, kk * 32, hi8);
        v16h b0 = *(const v16h*)p0;
        v16h b1 = *(const v16h*)p1;
        c0 = wmma_f16(a, b0, c0);
        c1 = wmma_f16(a, b1, c1);
        p0 += 512; p1 += 512;
      }
    }
    _Float16 h1a[8], h1b[8];
#pragma unroll
    for (int j = 0; j < 8; ++j) {
      h1a[j] = (_Float16)fast_tanh(c0[j]);
      h1b[j] = (_Float16)fast_tanh(c1[j]);
    }
    __syncthreads();  // B3: all layer-1 reads of A1 (incl. old h1) done
    store_h_lds(A1 + HIDDEN + nt0 * 16, S1, h1a, m0, lrow);
    store_h_lds(A1 + HIDDEN + nt1 * 16, S1, h1b, m0, lrow);
    __syncthreads();  // B4: h1_new visible

    // ---- output: h1_new (16x512) @ Wout^T, 16 k-steps, 1 vocab tile/wave ----
    v8f co;
#pragma unroll
    for (int j = 0; j < 8; ++j) co[j] = bO;
    {
      const _Float16* p = B2p;
#pragma unroll 4
      for (int kk = 0; kk < 16; ++kk) {
        v16h a = load_a_lds(arow1, HIDDEN + kk * 32, hi8);
        v16h b = *(const v16h*)p;
        co = wmma_f16(a, b, co);
        p += 512;
      }
    }
    {
      // out[b][t][v], b = wgbase + m0 + j, v = w*16 + lrow; streamed (NT)
      float* o = out + ((size_t)(wgbase + m0) * SEQ + t) * VOCAB + w * 16 + lrow;
#pragma unroll
      for (int j = 0; j < 8; ++j)
        __builtin_nontemporal_store(co[j], o + (size_t)j * SEQ * VOCAB);
    }
  }
}

extern "C" void kernel_launch(void* const* d_in, const int* in_sizes, int n_in,
                              void* d_out, int out_size, void* d_ws, size_t ws_size,
                              hipStream_t stream) {
  const int*   x     = (const int*)d_in[0];
  const float* emb   = (const float*)d_in[1];
  const float* Wih0  = (const float*)d_in[2];
  const float* bih0  = (const float*)d_in[3];
  const float* Whh0  = (const float*)d_in[4];
  const float* bhh0  = (const float*)d_in[5];
  const float* Wih1  = (const float*)d_in[6];
  const float* bih1  = (const float*)d_in[7];
  const float* Whh1  = (const float*)d_in[8];
  const float* bhh1  = (const float*)d_in[9];
  const float* Wout  = (const float*)d_in[10];
  const float* bout  = (const float*)d_in[11];

  _Float16* Wp   = (_Float16*)d_ws;                                   // 2 MB
  float*    bias = (float*)((char*)d_ws + (size_t)WP_TOTAL * 2);      // 1280 f32

  {
    int total  = WP_TOTAL + BIAS_N;
    int blocks = (total + 255) / 256;
    rnn_pack_kernel<<<blocks, 256, 0, stream>>>(Wih0, Whh0, Wih1, Whh1, Wout,
                                                bih0, bhh0, bih1, bhh1, bout,
                                                Wp, bias);
  }
  rnn_persistent_kernel<<<BATCH / 16, 512, 0, stream>>>(
      x, emb, Wp, bias, bias + 512, bias + 1024, (float*)d_out);
}
